// Attention_77472620085715
// MI455X (gfx1250) — matrix operand order repaired and checked by reading, not compiled, not run on hardware
//
#include <hip/hip_runtime.h>
#include <hip/hip_bf16.h>

// ---------------------------------------------------------------------------
// Attention forward for MI455X (gfx1250, wave32, WMMA).
//   qp = q@Wq.T+bq ; kp = k@Wk.T+bk ; vp = v@Wv.T+bv
//   P  = softmax(qp@kp.T)    (returned as output #2, so it must hit HBM)
//   out = (P@vp)@Wo.T + bo   (output #1)
// GEMMs: v_wmma_f32_16x16x32_bf16, f32 accumulate, 128x128 block tile,
// double-buffered LDS (one barrier per K-step). bf16->bf16 staging uses
// GLOBAL_LOAD_ASYNC_TO_LDS_B128 (ASYNCcnt) when the builtin is available.
// ---------------------------------------------------------------------------

typedef __attribute__((ext_vector_type(16))) __bf16 v16bf;
typedef __attribute__((ext_vector_type(8)))  __bf16 v8bf;
typedef __attribute__((ext_vector_type(8)))  float  v8f;

#define LDS_STRIDE 40   // 32 K-elems + 8 pad; 80B row pitch = 20 banks, gcd(20,64)=4
                        // -> 16 lane starts cover all 64 banks: conflict-free b128

#if defined(__has_builtin)
#if __has_builtin(__builtin_amdgcn_global_load_async_to_lds_b128)
#define HAVE_ASYNC_B128 1
#endif
#endif

#ifdef HAVE_ASYNC_B128
// Builtin signature (from clang diagnostic): arg0 = int4 __device__* (AS1),
// arg1 = int4 __shared__* (AS3), then two integer immediates (offset, cpol).
typedef int v4i __attribute__((vector_size(16)));
typedef __attribute__((address_space(1))) v4i as1_v4i;
typedef __attribute__((address_space(3))) v4i as3_v4i;

// AS1 address == flat address for global; flat address of __shared__ carries
// the LDS byte offset in its low 32 bits (per the ISA).
__device__ __forceinline__ void async_b128(const void* gsrc, void* ldst) {
  __builtin_amdgcn_global_load_async_to_lds_b128(
      (as1_v4i*)(unsigned long long)gsrc,
      (as3_v4i*)(unsigned int)(unsigned long long)ldst,
      0, 0);
}
#endif

__device__ __forceinline__ void async_wait0() {
#ifdef HAVE_ASYNC_B128
#if __has_builtin(__builtin_amdgcn_s_wait_asynccnt)
  __builtin_amdgcn_s_wait_asynccnt(0);
#else
  asm volatile("s_wait_asynccnt 0" ::: "memory");
#endif
#endif
}

__device__ __forceinline__ v16bf concat8(v8bf a, v8bf b) {
  return __builtin_shufflevector(a, b, 0,1,2,3,4,5,6,7,8,9,10,11,12,13,14,15);
}

// A fragment (16x32 bf16): lane = {h = lane>>4, m = lane&15}
// vector = A[m][h*16 .. +15]
__device__ __forceinline__ v16bf load_frag_a(const __bf16* base, int lane) {
  const int m = lane & 15, h = lane >> 4;
  const v8bf* p0 = (const v8bf*)(base + m * LDS_STRIDE + h * 16);
  const v8bf* p1 = (const v8bf*)(base + m * LDS_STRIDE + h * 16 + 8);
  return concat8(*p0, *p1);
}

// B fragment (32x16 bf16): lane = {h = lane>>4, n = lane&15}
// vector = B[h*16 .. +15][n]; B staged transposed in LDS as Bt[n][k],
// so 16 contiguous bf16 per lane.
__device__ __forceinline__ v16bf load_frag_b(const __bf16* base, int lane) {
  const int n = lane & 15, h = lane >> 4;
  const v8bf* p0 = (const v8bf*)(base + n * LDS_STRIDE + h * 16);
  const v8bf* p1 = (const v8bf*)(base + n * LDS_STRIDE + h * 16 + 8);
  return concat8(*p0, *p1);
}

// Stage 16 contiguous source elements into LDS as bf16 (overloads pick path).
__device__ __forceinline__ void stage16(__bf16* dst, const float* src) {
  const float4* s = (const float4*)src;           // 4x global_load_b128
  float4 a = s[0], b = s[1], c = s[2], d = s[3];
  v8bf lo = { (__bf16)a.x,(__bf16)a.y,(__bf16)a.z,(__bf16)a.w,
              (__bf16)b.x,(__bf16)b.y,(__bf16)b.z,(__bf16)b.w };
  v8bf hi = { (__bf16)c.x,(__bf16)c.y,(__bf16)c.z,(__bf16)c.w,
              (__bf16)d.x,(__bf16)d.y,(__bf16)d.z,(__bf16)d.w };
  *(v8bf*)dst = lo;  *(v8bf*)(dst + 8) = hi;
}
__device__ __forceinline__ void stage16(__bf16* dst, const __bf16* src) {
#ifdef HAVE_ASYNC_B128
  // 32B per thread via two async b128 copies straight into LDS (ASYNCcnt),
  // no VGPR round-trip.
  async_b128(src,     dst);
  async_b128(src + 8, dst + 8);
#else
  const v8bf* s = (const v8bf*)src;               // 2x global_load_b128
  *(v8bf*)dst = s[0];  *(v8bf*)(dst + 8) = s[1];
#endif
}

// C = A[MxK] * op(B) (+ bias), 128x128 block tile, 8 waves, wave tile 32x64.
// TRANSB=true : B is [N,K] row-major, C += A*B^T   (Linear / qk^T)
// TRANSB=false: B is [K,N] row-major, C += A*B     (P @ vp)
template <typename TA, typename TB, bool TRANSB, bool BIAS, bool OUT_BF16>
__global__ __launch_bounds__(256)
void gemm_bf16_wmma(const TA* __restrict__ A, const TB* __restrict__ B,
                    const float* __restrict__ bias, void* __restrict__ Cv,
                    int M, int N, int K, int lda, int ldb, int ldc)
{
  __shared__ __attribute__((aligned(16))) __bf16 sA[2][128 * LDS_STRIDE];
  __shared__ __attribute__((aligned(16))) __bf16 sB[2][128 * LDS_STRIDE]; // Bt[n][k]

  const int tid  = threadIdx.x;
  const int lane = tid & 31;
  const int wave = tid >> 5;
  const int wr   = wave & 3;   // wave row group (4)  -> rows wr*32..+31
  const int wc   = wave >> 2;  // wave col group (2)  -> cols wc*64..+63

  const long blockM = (long)blockIdx.y * 128;
  const long blockN = (long)blockIdx.x * 128;

  const int ldRow  = tid >> 1;         // 0..127
  const int ldColH = (tid & 1) * 16;   // 0 or 16

  auto stage_tile = [&](int buf, int k0) {
    // A tile: sA[m][k]
    stage16(&sA[buf][ldRow * LDS_STRIDE + ldColH],
            A + (blockM + ldRow) * (long)lda + k0 + ldColH);
    // B tile transposed: sB[n][k]
    if constexpr (TRANSB) {
      stage16(&sB[buf][ldRow * LDS_STRIDE + ldColH],
              B + (blockN + ldRow) * (long)ldb + k0 + ldColH);
    } else {
      // B[k][n]: read rows of B (coalesced over n), scatter into sB[n][k].
      const int kk = tid >> 3;           // 0..31
      const int nh = (tid & 7) * 16;     // 0..112
      const TB* gp = B + (long)(k0 + kk) * ldb + blockN + nh;
      const v8bf x0 = ((const v8bf*)gp)[0];
      const v8bf x1 = ((const v8bf*)gp)[1];
      #pragma unroll
      for (int i = 0; i < 8; ++i) sB[buf][(nh + i) * LDS_STRIDE + kk] = x0[i];
      #pragma unroll
      for (int i = 0; i < 8; ++i) sB[buf][(nh + 8 + i) * LDS_STRIDE + kk] = x1[i];
    }
  };

  v8f acc[2][4] = {};

  const int ksteps = K >> 5;            // K/32; all dims divide evenly here
  stage_tile(0, 0);
  async_wait0();
  __syncthreads();

  for (int s = 0; s < ksteps; ++s) {
    const int cur = s & 1;

    // Kick off next tile's global->LDS traffic before touching the matrix
    // pipe so it overlaps the WMMAs (double buffer -> one barrier/K-step).
    if (s + 1 < ksteps) stage_tile(cur ^ 1, (s + 1) << 5);
    if (s + 2 < ksteps) {   // keep L2 warm two tiles ahead (global_prefetch_b8)
      __builtin_prefetch(A + (blockM + ldRow) * (long)lda + ((s + 2) << 5) + ldColH, 0, 1);
      if constexpr (TRANSB)
        __builtin_prefetch(B + (blockN + ldRow) * (long)ldb + ((s + 2) << 5) + ldColH, 0, 1);
      else
        __builtin_prefetch(B + (long)(((s + 2) << 5) + (tid >> 3)) * ldb + blockN + (tid & 7) * 16, 0, 1);
    }

    v16bf af[2];
    #pragma unroll
    for (int mt = 0; mt < 2; ++mt)
      af[mt] = load_frag_a(&sA[cur][(wr * 32 + mt * 16) * LDS_STRIDE], lane);
    v16bf bfr[4];
    #pragma unroll
    for (int nt = 0; nt < 4; ++nt)
      bfr[nt] = load_frag_b(&sB[cur][(wc * 64 + nt * 16) * LDS_STRIDE], lane);

    #pragma unroll
    for (int mt = 0; mt < 2; ++mt)
      #pragma unroll
      for (int nt = 0; nt < 4; ++nt)
        acc[mt][nt] = __builtin_amdgcn_wmma_f32_16x16x32_bf16(
            false, af[mt], false, bfr[nt], (short)0, acc[mt][nt], false, false);

    async_wait0();          // staged-ahead async copies must land before signal
    __syncthreads();
  }

  // ---- epilogue: C/D layout (ISA 7.12.2): m = r + 8*(lane>>4), n = lane&15 -
  const int halfc = lane >> 4;
  const int nl    = lane & 15;
  #pragma unroll
  for (int mt = 0; mt < 2; ++mt) {
    #pragma unroll
    for (int nt = 0; nt < 4; ++nt) {
      const long gn = blockN + wc * 64 + nt * 16 + nl;
      const float bv = BIAS ? bias[gn] : 0.0f;
      #pragma unroll
      for (int r = 0; r < 8; ++r) {
        const long gm = blockM + wr * 32 + mt * 16 + r + halfc * 8;
        const float val = acc[mt][nt][r] + bv;
        if constexpr (OUT_BF16)
          ((__bf16*)Cv)[gm * (long)ldc + gn] = (__bf16)val;
        else
          ((float*)Cv)[gm * (long)ldc + gn] = val;
      }
    }
  }
}

// Row softmax, in place: one block per row, row (32KB) cached in LDS.
__global__ __launch_bounds__(256)
void softmax_rows(float* __restrict__ scores, int n)
{
  __shared__ __attribute__((aligned(16))) float row[8192];
  __shared__ float red[256];
  const int t = threadIdx.x;
  float4* __restrict__ g = (float4*)(scores + (size_t)blockIdx.x * n);
  float4* l = (float4*)row;

  float lmax = -3.402823466e38f;
  #pragma unroll
  for (int i = 0; i < 8; ++i) {
    float4 v = g[i * 256 + t];
    l[i * 256 + t] = v;
    lmax = fmaxf(lmax, fmaxf(fmaxf(v.x, v.y), fmaxf(v.z, v.w)));
  }
  red[t] = lmax;
  __syncthreads();
  for (int s = 128; s > 0; s >>= 1) {
    if (t < s) red[t] = fmaxf(red[t], red[t + s]);
    __syncthreads();
  }
  const float m = red[0];
  __syncthreads();

  float lsum = 0.0f;
  #pragma unroll
  for (int i = 0; i < 8; ++i) {
    float4 v = l[i * 256 + t];
    v.x = __expf(v.x - m); v.y = __expf(v.y - m);
    v.z = __expf(v.z - m); v.w = __expf(v.w - m);
    l[i * 256 + t] = v;
    lsum += v.x + v.y + v.z + v.w;
  }
  red[t] = lsum;
  __syncthreads();
  for (int s = 128; s > 0; s >>= 1) {
    if (t < s) red[t] += red[t + s];
    __syncthreads();
  }
  const float inv = 1.0f / red[0];
  #pragma unroll
  for (int i = 0; i < 8; ++i) {
    float4 v = l[i * 256 + t];
    v.x *= inv; v.y *= inv; v.z *= inv; v.w *= inv;
    g[i * 256 + t] = v;
  }
}

extern "C" void kernel_launch(void* const* d_in, const int* in_sizes, int n_in,
                              void* d_out, int out_size, void* d_ws, size_t ws_size,
                              hipStream_t stream)
{
  (void)in_sizes; (void)n_in; (void)out_size; (void)ws_size;
  constexpr int N = 8192, H = 512;

  const float* q  = (const float*)d_in[0];
  const float* k  = (const float*)d_in[1];
  const float* v  = (const float*)d_in[2];
  const float* Wq = (const float*)d_in[3];
  const float* bq = (const float*)d_in[4];
  const float* Wk = (const float*)d_in[5];
  const float* bk = (const float*)d_in[6];
  const float* Wv = (const float*)d_in[7];
  const float* bv = (const float*)d_in[8];
  const float* Wo = (const float*)d_in[9];
  const float* bo = (const float*)d_in[10];

  float* out   = (float*)d_out;                 // [N,H]
  float* probs = out + (size_t)N * H;           // [N,N] (returned normalized)

  __bf16* qp  = (__bf16*)d_ws;                  // 4 x N*H bf16 = 32 MB scratch
  __bf16* kp  = qp + (size_t)N * H;
  __bf16* vp  = kp + (size_t)N * H;
  __bf16* att = vp + (size_t)N * H;

  const dim3 blk(256);
  const dim3 gProj(H / 128, N / 128);           // (4, 64)
  const dim3 gScore(N / 128, N / 128);          // (64, 64)

  // 1) projections: x @ W.T + b  -> bf16
  gemm_bf16_wmma<float, float, true, true, true>
      <<<gProj, blk, 0, stream>>>(q, Wq, bq, qp, N, H, H, H, H, H);
  gemm_bf16_wmma<float, float, true, true, true>
      <<<gProj, blk, 0, stream>>>(k, Wk, bk, kp, N, H, H, H, H, H);
  gemm_bf16_wmma<float, float, true, true, true>
      <<<gProj, blk, 0, stream>>>(v, Wv, bv, vp, N, H, H, H, H, H);

  // 2) scores = qp @ kp.T -> fp32 into d_out's normalized region
  //    (both operands bf16 -> async LDS staging path)
  gemm_bf16_wmma<__bf16, __bf16, true, false, false>
      <<<gScore, blk, 0, stream>>>(qp, kp, nullptr, probs, N, N, H, H, H, N);

  // 3) row softmax, in place (this IS the second reference output)
  softmax_rows<<<N, 256, 0, stream>>>(probs, N);

  // 4) att = P @ vp (A fp32 from d_out, converted while staging) -> bf16
  gemm_bf16_wmma<float, __bf16, false, false, true>
      <<<gProj, blk, 0, stream>>>(probs, vp, nullptr, att, N, H, N, N, H, H);

  // 5) out = att @ Wo.T + bo -> fp32
  gemm_bf16_wmma<__bf16, float, true, true, false>
      <<<gProj, blk, 0, stream>>>(att, Wo, bo, out, N, H, H, H, H, H);
}
